// LSTMAttention_47760036332051
// MI455X (gfx1250) — compile-verified, matrix operand order checked by reading, not run on hardware
//
#include <hip/hip_runtime.h>
#include <cstddef>
#include <cstdint>

// ---------------------------------------------------------------------------
// Types for CDNA5 WMMA (wave32) and TDM descriptors
// ---------------------------------------------------------------------------
typedef __bf16 bf16;
typedef __attribute__((ext_vector_type(16))) bf16     v16bf;
typedef __attribute__((ext_vector_type(8)))  bf16     v8bf;   // one 16B (b128) chunk
typedef __attribute__((ext_vector_type(8)))  float    v8f;    // 16x16 f32 accumulator
typedef __attribute__((ext_vector_type(4)))  unsigned v4u;    // TDM D# group 0
typedef __attribute__((ext_vector_type(8)))  int      v8i;    // TDM D# group 1 / spare
typedef __attribute__((ext_vector_type(4)))  int      v4i;    // TDM D# groups 2/3

#define BB   16
#define LL   1024
#define INC  512
#define HH   1024
#define OO   256
#define G4H  4096

static __device__ __forceinline__ bf16 f2bf(float f) {
    union { float f; unsigned u; } v; v.f = f;
    unsigned r = v.u + 0x7FFFu + ((v.u >> 16) & 1u);   // round-to-nearest-even
    union { unsigned short s; bf16 b; } o; o.s = (unsigned short)(r >> 16);
    return o.b;
}
static __device__ __forceinline__ float bf2f(bf16 x) {
    union { bf16 b; unsigned short s; } u; u.b = x;
    union { unsigned u32; float f; } o; o.u32 = ((unsigned)u.s) << 16;
    return o.f;
}
static __device__ __forceinline__ float sigm(float x) { return 1.0f / (1.0f + __expf(-x)); }

// ---------------------------------------------------------------------------
// fp32 -> bf16 conversion (weights and x)
// ---------------------------------------------------------------------------
__global__ void cvt_kernel(const float* __restrict__ in, bf16* __restrict__ out, int n) {
    int i = blockIdx.x * blockDim.x + threadIdx.x;
    if (i < n) out[i] = f2bf(in[i]);
}

// ---------------------------------------------------------------------------
// Generic WMMA GEMM: C[M,N] = A[M,K] (bf16 row-major) * W[N,K]^T (bf16 row-major)
// 8 waves/block, each wave owns a 16(M) x 32(N) tile -> block tile 16 x 256.
// EP 0: C_f32 = acc + bias
// EP 1: C_bf16 = tanh(acc + bias)
// EP 2: C_f32 = sigmoid(acc + bias) + item[b][n]   (b = row/L)
// ---------------------------------------------------------------------------
template<int EP>
__global__ __launch_bounds__(256)
void gemm_wmma(const bf16* __restrict__ A, int lda,
               const bf16* __restrict__ W, int ldw,
               const float* __restrict__ bias,
               float* __restrict__ Cf, bf16* __restrict__ Cb, int ldc,
               const float* __restrict__ item, int K)
{
    const int lane = threadIdx.x & 31;
    const int wv   = threadIdx.x >> 5;
    const int r    = lane & 15;
    const int hf   = lane >> 4;
    const int m0   = blockIdx.x * 16;
    const int cb   = blockIdx.y * 256 + wv * 32;

    const bf16* Ap  = A + (size_t)(m0 + r) * lda + hf * 8;
    const bf16* Bp0 = W + (size_t)(cb + r) * ldw + hf * 8;
    const bf16* Bp1 = W + (size_t)(cb + 16 + r) * ldw + hf * 8;

    v8f acc0 = {}; v8f acc1 = {};
    const int kt = K >> 5;
    for (int kk = 0; kk < kt; ++kk) {
        union { v16bf v; v8bf h[2]; } a, b0, b1;
        a.h[0]  = *(const v8bf*)(Ap  + kk * 32);
        a.h[1]  = *(const v8bf*)(Ap  + kk * 32 + 16);
        b0.h[0] = *(const v8bf*)(Bp0 + kk * 32);
        b0.h[1] = *(const v8bf*)(Bp0 + kk * 32 + 16);
        b1.h[0] = *(const v8bf*)(Bp1 + kk * 32);
        b1.h[1] = *(const v8bf*)(Bp1 + kk * 32 + 16);
        __builtin_prefetch(Ap  + kk * 32 + 128, 0, 1);   // global_prefetch_b8
        __builtin_prefetch(Bp0 + kk * 32 + 128, 0, 1);
        acc0 = __builtin_amdgcn_wmma_f32_16x16x32_bf16(false, a.v, false, b0.v,
                                                       (short)0, acc0, false, false);
        acc1 = __builtin_amdgcn_wmma_f32_16x16x32_bf16(false, a.v, false, b1.v,
                                                       (short)0, acc1, false, false);
    }

    for (int v = 0; v < 8; ++v) {
        const int m = hf * 8 + v;
        const size_t row = (size_t)(m0 + m) * ldc;
        {
            const int n = cb + r;
            float x = acc0[v] + bias[n];
            if (EP == 0) Cf[row + n] = x;
            if (EP == 1) Cb[row + n] = f2bf(tanhf(x));
            if (EP == 2) { int b = (m0 + m) >> 10; Cf[row + n] = sigm(x) + item[b * OO + n]; }
        }
        {
            const int n = cb + 16 + r;
            float x = acc1[v] + bias[n];
            if (EP == 0) Cf[row + n] = x;
            if (EP == 1) Cb[row + n] = f2bf(tanhf(x));
            if (EP == 2) { int b = (m0 + m) >> 10; Cf[row + n] = sigm(x) + item[b * OO + n]; }
        }
    }
}

// ---------------------------------------------------------------------------
// Grid-wide barrier: monotone arrive counter (memset to 0 once per launch).
// ---------------------------------------------------------------------------
static __device__ __forceinline__ void gsync(unsigned* cnt, unsigned target) {
    __syncthreads();
    if (threadIdx.x == 0) {
        __threadfence();
        atomicAdd(cnt, 1u);
        while (atomicAdd(cnt, 0u) < target) __builtin_amdgcn_s_sleep(8);
        __threadfence();
    }
    __syncthreads();
}

// ---------------------------------------------------------------------------
// Persistent LSTM scan. 64 WGs x 128 threads (4 waves). WG g owns h-columns
// [g*16, g*16+16); wave q owns gate q (i,f,g,o). Wave q's 16 Whh rows x K=1024
// (bf16) are staged into LDS by ONE Tensor-Data-Mover descriptor and stay
// resident for all 1024 steps. TDM pads the LDS destination 16B per 1024B so
// every 16B B-fragment stays b128-aligned and rows rotate across LDS banks.
// Per step: 32 x v_wmma_f32_16x16x32_bf16 per wave, fused gate epilogue,
// double-buffered h + one grid barrier.
// ---------------------------------------------------------------------------
__global__ __launch_bounds__(128)
void lstm_scan(const bf16* __restrict__ Whh, const float* __restrict__ bhh,
               const float* __restrict__ xg, bf16* __restrict__ hbuf,
               bf16* __restrict__ out, unsigned* __restrict__ cnt)
{
    constexpr int NWG   = 64;
    constexpr int WROW  = 1040;                    // 2048B data + 2x16B TDM pad
    __shared__ __align__(16) bf16 sW[4 * 16 * WROW + 64];  // ~133 KB of 320 KB WGP LDS
    __shared__ float sG[4 * 16 * 16];                      // gate pre-activations
    __shared__ float sC[16 * 16];                          // cell state

    const int g    = blockIdx.x;
    const int tid  = threadIdx.x;
    const int lane = tid & 31;
    const int q    = tid >> 5;                     // gate index 0..3
    const int r    = lane & 15;
    const int hf   = lane >> 4;
    const int g16  = g * 16;

    // ---- TDM: stage wave q's 16x1024 bf16 Whh slice into LDS --------------
    {
        // D# group 0: count=1 | lds_addr | 57b global_addr | type=2
        unsigned ldsBase = (unsigned)(size_t)(&sW[0]) + (unsigned)(q * 16 * 2080);
        unsigned long long ga =
            (unsigned long long)(size_t)&Whh[(size_t)(q * 1024 + g16) * 1024];
        v4u g0;
        g0[0] = 1u;                                       // count=1 (valid user D#)
        g0[1] = ldsBase;                                  // bits 63:32 = lds_addr
        g0[2] = (unsigned)(ga & 0xFFFFFFFFu);             // global_addr[31:0]
        g0[3] = (unsigned)((ga >> 32) & 0x1FFFFFFu) | (2u << 30);  // addr[56:32]|type=2
        // D# group 1: data_size=8B, pad_enable, pad_interval=1024B(code7),
        //             pad_amount=16B(code3); dims/strides in 8B units.
        v8i g1;
        g1[0] = (3 << 16) | (1 << 20) | (7 << 22) | (3 << 25);
        g1[1] = (256 << 16);                              // tensor_dim0 = 256 (2048B row)
        g1[2] = (16 << 16);                               // tensor_dim1 = 16 rows
        g1[3] = (256 << 16);                              // tile_dim0 = 256
        g1[4] = 16;                                       // tile_dim1 = 16, tile_dim2 = 0
        g1[5] = 256;                                      // tensor_dim0_stride = 2048B
        g1[6] = 0;
        g1[7] = 0;
        v4i zz = {0, 0, 0, 0};                            // groups 2/3 unused (2D tile)
        v8i z8 = {0, 0, 0, 0, 0, 0, 0, 0};                // trailing group (clang-23 form)
        __builtin_amdgcn_tensor_load_to_lds(g0, g1, zz, zz, z8, 0);  // tensor_load_to_lds
    }

    // h0 = 0 (own slice), c0 = 0.
    const bf16 z = f2bf(0.0f);
    for (int p = tid; p < 256; p += 128) {
        int b = p >> 4, j = p & 15;
        hbuf[(size_t)b * 1024 + g16 + j] = z;
        sC[p] = 0.0f;
    }
    __builtin_amdgcn_s_wait_tensorcnt(0);          // s_wait_tensorcnt 0x0
    __syncthreads();
    gsync(cnt, NWG);                               // all h0 slices visible

    const int   gcol = q * 1024 + g16 + r;         // this lane's gate column
    const float bias = bhh[gcol];
    const bf16* Brow = &sW[(q * 16 + r) * WROW];   // row footprint 2080B

    for (int t = 0; t < LL; ++t) {
        const int cur = t & 1, nxt = cur ^ 1;
        const bf16* h  = hbuf + (size_t)cur * (16 * 1024);
        const bf16* Ap = h + (size_t)r * 1024 + hf * 8;

        v8f acc = {};
        for (int kk = 0; kk < 32; ++kk) {
            const int kb  = kk * 32 + hf * 8;
            const int kbp = kb + ((kb >> 9) << 3);         // +8 elems (16B) per 1024B pad
            union { v16bf v; v8bf c[2]; } a, b;
            a.c[0] = *(const v8bf*)(Ap + kk * 32);
            a.c[1] = *(const v8bf*)(Ap + kk * 32 + 16);
            b.c[0] = *(const v8bf*)(Brow + kbp);           // ds_load_b128
            b.c[1] = *(const v8bf*)(Brow + kbp + 16);
            acc = __builtin_amdgcn_wmma_f32_16x16x32_bf16(false, a.v, false, b.v,
                                                          (short)0, acc, false, false);
        }
        // gates = wmma + xg[:,t,:] + bhh  -> LDS
        for (int v = 0; v < 8; ++v) {
            int m = hf * 8 + v;                    // batch row
            float x = acc[v] + xg[(size_t)m * (LL * G4H) + (size_t)t * G4H + gcol] + bias;
            sG[(q * 16 + m) * 16 + r] = x;
        }
        __syncthreads();
        // pointwise LSTM update: each thread owns 2 (batch, h-col) cells
        for (int p = tid; p < 256; p += 128) {
            int b = p >> 4, j = p & 15;
            float gi = sigm (sG[(0 * 16 + b) * 16 + j]);
            float gf = sigm (sG[(1 * 16 + b) * 16 + j]);
            float gg = tanhf(sG[(2 * 16 + b) * 16 + j]);
            float go = sigm (sG[(3 * 16 + b) * 16 + j]);
            float c  = gf * sC[p] + gi * gg;
            sC[p] = c;
            bf16 hb = f2bf(go * tanhf(c));
            hbuf[(size_t)nxt * (16 * 1024) + (size_t)b * 1024 + g16 + j] = hb;
            out[((size_t)b * LL + t) * HH + g16 + j] = hb;
        }
        gsync(cnt, (unsigned)NWG * (t + 2));
    }
}

// ---------------------------------------------------------------------------
// score[b,t] = s[b,t,:] . v_w + v_b          (one wave per row)
// ---------------------------------------------------------------------------
__global__ __launch_bounds__(256)
void score_kernel(const bf16* __restrict__ s, const bf16* __restrict__ vw,
                  const float* __restrict__ vb, float* __restrict__ score)
{
    int row  = blockIdx.x * 8 + (threadIdx.x >> 5);
    int lane = threadIdx.x & 31;
    const bf16* sp = s + (size_t)row * HH;
    float acc = 0.0f;
    for (int k = lane; k < HH; k += 32) acc += bf2f(sp[k]) * bf2f(vw[k]);
    for (int off = 16; off; off >>= 1) acc += __shfl_xor(acc, off, 32);
    if (lane == 0) score[row] = acc + vb[0];
}

// ---------------------------------------------------------------------------
// ctx[b,h] = sum_t score[b,t] * out[b,t,h]
// ---------------------------------------------------------------------------
__global__ __launch_bounds__(256)
void ctx_kernel(const float* __restrict__ score, const bf16* __restrict__ out1,
                float* __restrict__ ctx)
{
    int b = blockIdx.x;
    int h = blockIdx.y * 256 + threadIdx.x;
    float acc = 0.0f;
    for (int t = 0; t < LL; ++t)
        acc += score[b * LL + t] * bf2f(out1[((size_t)b * LL + t) * HH + h]);
    ctx[b * HH + h] = acc;
}

// ---------------------------------------------------------------------------
// item[b,o] = sigmoid(ctx[b,:] . fc_w[o,:] + fc_b[o])   (one wave per output)
// ---------------------------------------------------------------------------
__global__ __launch_bounds__(256)
void item_kernel(const float* __restrict__ ctx, const float* __restrict__ fcw,
                 const float* __restrict__ fcb, float* __restrict__ item)
{
    int idx  = blockIdx.x * 8 + (threadIdx.x >> 5);   // 0..4095
    int lane = threadIdx.x & 31;
    int b = idx >> 8, o = idx & 255;
    float acc = 0.0f;
    for (int k = lane; k < HH; k += 32) acc += ctx[b * HH + k] * fcw[o * HH + k];
    for (int off = 16; off; off >>= 1) acc += __shfl_xor(acc, off, 32);
    if (lane == 0) item[idx] = sigm(acc + fcb[o]);
}

// ---------------------------------------------------------------------------
extern "C" void kernel_launch(void* const* d_in, const int* in_sizes, int n_in,
                              void* d_out, int out_size, void* d_ws, size_t ws_size,
                              hipStream_t stream)
{
    const float* x    = (const float*)d_in[0];
    const float* Wih0 = (const float*)d_in[1];
    const float* Whh0 = (const float*)d_in[2];
    const float* bih0 = (const float*)d_in[3];
    const float* bhh0 = (const float*)d_in[4];
    const float* Wih1 = (const float*)d_in[5];
    const float* Whh1 = (const float*)d_in[6];
    const float* bih1 = (const float*)d_in[7];
    const float* bhh1 = (const float*)d_in[8];
    const float* fc_w = (const float*)d_in[9];
    const float* fc_b = (const float*)d_in[10];
    const float* wh_w = (const float*)d_in[11];
    const float* wh_b = (const float*)d_in[12];
    const float* v_w  = (const float*)d_in[13];
    const float* v_b  = (const float*)d_in[14];
    float* outp = (float*)d_out;

    char* ws = (char*)d_ws;
    size_t off = 0;
    auto alloc = [&](size_t bytes) -> void* {
        void* p = ws + off;
        off = (off + bytes + 255) & ~(size_t)255;
        return p;
    };

    bf16*  x_bf    = (bf16*)alloc((size_t)BB * LL * INC * 2);
    bf16*  wih0_bf = (bf16*)alloc((size_t)G4H * INC * 2);
    bf16*  whh0_bf = (bf16*)alloc((size_t)G4H * HH * 2);
    bf16*  wih1_bf = (bf16*)alloc((size_t)G4H * HH * 2);
    bf16*  whh1_bf = (bf16*)alloc((size_t)G4H * HH * 2);
    bf16*  whw_bf  = (bf16*)alloc((size_t)HH * HH * 2);
    bf16*  fcw_bf  = (bf16*)alloc((size_t)OO * HH * 2);
    bf16*  vw_bf   = (bf16*)alloc((size_t)HH * 2);
    float* xg      = (float*)alloc((size_t)BB * LL * G4H * 4);   // 256 MB, reused
    bf16*  out0_bf = (bf16*)alloc((size_t)BB * LL * HH * 2);
    bf16*  out1_bf = (bf16*)alloc((size_t)BB * LL * HH * 2);
    float* score   = (float*)alloc((size_t)BB * LL * 4);
    float* ctxb    = (float*)alloc((size_t)BB * HH * 4);
    float* itemb   = (float*)alloc((size_t)BB * OO * 4);
    bf16*  hbuf    = (bf16*)alloc((size_t)2 * BB * HH * 2);
    unsigned* cnt  = (unsigned*)alloc(256);
    bf16*  s_bf    = (bf16*)xg;   // attention scores tensor aliases xg (free by then)

    (void)hipMemsetAsync(cnt, 0, 2 * sizeof(unsigned), stream);

    auto cvt = [&](const float* in, bf16* ob, int n) {
        cvt_kernel<<<(n + 255) / 256, 256, 0, stream>>>(in, ob, n);
    };
    cvt(x,    x_bf,    BB * LL * INC);
    cvt(Wih0, wih0_bf, G4H * INC);
    cvt(Whh0, whh0_bf, G4H * HH);
    cvt(Wih1, wih1_bf, G4H * HH);
    cvt(Whh1, whh1_bf, G4H * HH);
    cvt(wh_w, whw_bf,  HH * HH);
    cvt(fc_w, fcw_bf,  OO * HH);
    cvt(v_w,  vw_bf,   HH);

    dim3 blk(256);
    // layer 0: xg = x @ Wih0^T + bih0
    gemm_wmma<0><<<dim3(1024, 16), blk, 0, stream>>>(x_bf, INC, wih0_bf, INC, bih0,
                                                     xg, nullptr, G4H, nullptr, INC);
    lstm_scan<<<64, 128, 0, stream>>>(whh0_bf, bhh0, xg, hbuf, out0_bf, cnt);
    // layer 1: xg = out0 @ Wih1^T + bih1
    gemm_wmma<0><<<dim3(1024, 16), blk, 0, stream>>>(out0_bf, HH, wih1_bf, HH, bih1,
                                                     xg, nullptr, G4H, nullptr, HH);
    lstm_scan<<<64, 128, 0, stream>>>(whh1_bf, bhh1, xg, hbuf, out1_bf, cnt + 1);

    // attention: s = tanh(out1 @ wh^T + wh_b)   (bf16, aliased into xg)
    gemm_wmma<1><<<dim3(1024, 4), blk, 0, stream>>>(out1_bf, HH, whw_bf, HH, wh_b,
                                                    nullptr, s_bf, HH, nullptr, HH);
    score_kernel<<<2048, 256, 0, stream>>>(s_bf, vw_bf, v_b, score);
    ctx_kernel<<<dim3(16, 4), 256, 0, stream>>>(score, out1_bf, ctxb);
    item_kernel<<<512, 256, 0, stream>>>(ctxb, fc_w, fc_b, itemb);

    // out = sigmoid(out1 @ fc^T + fc_b) + item   (fused epilogue)
    gemm_wmma<2><<<dim3(1024, 1), blk, 0, stream>>>(out1_bf, HH, fcw_bf, HH, fc_b,
                                                    outp, nullptr, OO, itemb, HH);
}
